// DotProductAttention_7421703487951
// MI455X (gfx1250) — matrix operand order repaired and checked by reading, not compiled, not run on hardware
//
#include <hip/hip_runtime.h>
#include <stdint.h>

// DotProductAttention for B=32,H=8,S=1024,D=16 on gfx1250 (MI455X).
// logits^T = K_tile x (0.25*Q)^T via v_wmma_f32_16x16x32_f16 (D=16 zero-padded)
// p = exp2(tanh(x)*10*log2e + bias), bias = mask ? -inf : 0   (|logit|<=10 -> no max pass)
// out^T += V^T x P^T via second wmma; P^T C-layout -> B-layout via 4 shfl_xor(16).
// A operands carry the K=16..31 zeros, so B operands need no lane-half zeroing.

typedef __attribute__((ext_vector_type(16))) _Float16 v16h;
typedef __attribute__((ext_vector_type(8)))  float    v8f;

#define S_LEN 1024
#define DH    16
#define KTILES 64      // S / 16
#define QTPW   4       // query tiles per wave
#define WPB    8       // waves per block

union V16 { v16h v; _Float16 h[16]; uint32_t u[8]; };

__device__ __forceinline__ uint32_t pk2(float a, float b) {
  // packs a -> [15:0], b -> [31:16] as f16 (RTZ)
  auto p = __builtin_amdgcn_cvt_pkrtz(a, b);
  uint32_t u;
  __builtin_memcpy(&u, &p, 4);
  return u;
}

__device__ __forceinline__ float dtanh(float x) {
#if __has_builtin(__builtin_amdgcn_tanhf)
  return __builtin_amdgcn_tanhf(x);
#else
  float e = __expf(2.0f * x);
  return 1.0f - 2.0f / (e + 1.0f);
#endif
}

__device__ __forceinline__ float dexp2(float x) {
#if __has_builtin(__builtin_amdgcn_exp2f)
  return __builtin_amdgcn_exp2f(x);
#else
  return exp2f(x);
#endif
}

__global__ __launch_bounds__(WPB * 32)
void attn_tanh_kernel(const float* __restrict__ Q,
                      const float* __restrict__ K,
                      const float* __restrict__ V,
                      const unsigned char* __restrict__ mask,
                      float* __restrict__ out) {
  const int lane = threadIdx.x & 31;
  const int warp = threadIdx.x >> 5;
  const int hi   = lane >> 4;        // lane half: 0 or 1
  const int lq   = lane & 15;

  const int wid = blockIdx.x * WPB + warp;   // 0..4095
  const int bh  = wid >> 4;                  // (b*H + h), 16 wave-groups per bh
  const int qg  = wid & 15;                  // wave-group within bh
  const int b   = bh >> 3;

  const size_t base = (size_t)bh * (S_LEN * DH);
  const float* Qb = Q + base;
  const float* Kb = K + base;
  const float* Vb = V + base;
  float*       Ob = out + base;
  const unsigned char* Mb = mask + (size_t)b * S_LEN;

  const float NEGINF = -__builtin_inff();
  const float C2 = 14.42695040888963f;   // 10 * log2(e)

  // ---- Hoist (0.25*Q)^T as the B operand of WMMA1: (K=dim, N=query) ----
  // N=lq, elems 0..7 = 0.25*Q(q, hi*8..hi*8+7); elems 8..15 = the other dims
  // (finite); they map to K=16..31 which multiply the zero half of aK.
  V16 bQ[QTPW];
#pragma unroll
  for (int t = 0; t < QTPW; ++t) {
    const float4* qp =
        (const float4*)(Qb + ((size_t)((qg * QTPW + t) * 16 + lq)) * DH);
#pragma unroll
    for (int j = 0; j < 4; ++j) {
      float4 f = qp[(j + 2 * hi) & 3];
      bQ[t].h[4 * j + 0] = (_Float16)(0.25f * f.x);
      bQ[t].h[4 * j + 1] = (_Float16)(0.25f * f.y);
      bQ[t].h[4 * j + 2] = (_Float16)(0.25f * f.z);
      bQ[t].h[4 * j + 3] = (_Float16)(0.25f * f.w);
    }
  }

  v8f zero;
#pragma unroll
  for (int i = 0; i < 8; ++i) zero[i] = 0.0f;

  v8f   O[QTPW];     // out^T accumulator: (M=dim, N=query) C-layout
  float rs[QTPW];    // per-lane partial row sums (keys hi*8..hi*8+7 of each tile)
#pragma unroll
  for (int t = 0; t < QTPW; ++t) { O[t] = zero; rs[t] = 0.0f; }

  for (int kt = 0; kt < KTILES; ++kt) {
    // ---- K tile as A operand of WMMA1: (M=key, K=dim) ----
    // lane: M=lq, elems 0..7 = dims hi*8..hi*8+7 ; elems 8..15 = 0 (K pad)
    V16 aK;
    {
      const float4* kp =
          (const float4*)(Kb + ((size_t)(kt * 16 + lq)) * DH + hi * 8);
      float4 k0 = kp[0], k1 = kp[1];
      aK.h[0] = (_Float16)k0.x; aK.h[1] = (_Float16)k0.y;
      aK.h[2] = (_Float16)k0.z; aK.h[3] = (_Float16)k0.w;
      aK.h[4] = (_Float16)k1.x; aK.h[5] = (_Float16)k1.y;
      aK.h[6] = (_Float16)k1.z; aK.h[7] = (_Float16)k1.w;
#pragma unroll
      for (int i = 8; i < 16; ++i) aK.h[i] = (_Float16)0.0f;
    }

    // ---- V^T tile as A operand of WMMA2: (M=dim, K=key) ----
    // lane: M=lq, elems 0..7 = keys kt*16 + hi*8 + i ; elems 8..15 = 0
    V16 aV;
#pragma unroll
    for (int i = 0; i < 8; ++i) {
      float v = Vb[((size_t)(kt * 16 + hi * 8 + i)) * DH + lq];
      aV.h[i] = (_Float16)v;
    }
#pragma unroll
    for (int i = 8; i < 16; ++i) aV.h[i] = (_Float16)0.0f;

    // ---- additive mask bias for this key tile (lane-invariant 16B load) ----
    // bias[j] corresponds to key kt*16 + hi*8 + j
    const uint4 m4 = *(const uint4*)(Mb + kt * 16);
    const uint32_t wA = hi ? m4.z : m4.x;  // bytes (hi*8)..(hi*8+3)
    const uint32_t wB = hi ? m4.w : m4.y;  // bytes (hi*8+4)..(hi*8+7)
    float bias[8];
#pragma unroll
    for (int j = 0; j < 8; ++j) {
      uint32_t mb = ((j < 4 ? wA : wB) >> ((j & 3) * 8)) & 0xffu;
      bias[j] = mb ? NEGINF : 0.0f;
    }

#pragma unroll
    for (int t = 0; t < QTPW; ++t) {
      // logits^T/4 = K_tile x (Q/4)^T -> C layout (M=key, N=query); lane holds
      // query lq, VGPR j = key kt*16 + hi*8 + j
      v8f lg = __builtin_amdgcn_wmma_f32_16x16x32_f16(
          false, aK.v, false, bQ[t].v, (short)0, zero, false, false);

      // p = exp2( tanh(x) * 10*log2e + bias )  ->  exactly 0 when masked
      float p[8];
#pragma unroll
      for (int j = 0; j < 8; ++j) {
        float th = dtanh(lg[j]);
        p[j] = dexp2(__builtin_fmaf(th, C2, bias[j]));
        rs[t] += p[j];
      }

      // ---- build P^T as B operand of WMMA2: (K=key, N=query) ----
      // lanes 0-15: u[0..3] = own keys 0..7 (packed f16), u[4..7] = keys 8..15
      // fetched from lane+16.  Lanes 16-31: u[0..3] = own keys 8..15, u[4..7] = keys 0..7 from
      // lane-16.  In all lanes u[4..7] (K=16..31) hit the zero half of aV -> contribute 0.
      uint32_t pk[4], ot[4];
#pragma unroll
      for (int j = 0; j < 4; ++j) pk[j] = pk2(p[2 * j], p[2 * j + 1]);
#pragma unroll
      for (int j = 0; j < 4; ++j) ot[j] = __shfl_xor(pk[j], 16, 32);
      V16 bP;
#pragma unroll
      for (int j = 0; j < 4; ++j) {
        bP.u[j]     = pk[j];
        bP.u[4 + j] = ot[j];
      }

      // out^T += V^T x P^T
      O[t] = __builtin_amdgcn_wmma_f32_16x16x32_f16(
          false, aV.v, false, bP.v, (short)0, O[t], false, false);
    }
  }

  // ---- normalize and store ----
  // O C-layout: lane holds query lq, VGPR j = dim hi*8 + j -> 2x b128 stores
#pragma unroll
  for (int t = 0; t < QTPW; ++t) {
    float tot = rs[t] + __shfl_xor(rs[t], 16, 32);
    float rinv = __builtin_amdgcn_rcpf(tot);
    float* op = Ob + ((size_t)((qg * QTPW + t) * 16 + lq)) * DH + hi * 8;
    float4 o0, o1;
    o0.x = O[t][0] * rinv; o0.y = O[t][1] * rinv;
    o0.z = O[t][2] * rinv; o0.w = O[t][3] * rinv;
    o1.x = O[t][4] * rinv; o1.y = O[t][5] * rinv;
    o1.z = O[t][6] * rinv; o1.w = O[t][7] * rinv;
    ((float4*)op)[0] = o0;
    ((float4*)op)[1] = o1;
  }
}

extern "C" void kernel_launch(void* const* d_in, const int* in_sizes, int n_in,
                              void* d_out, int out_size, void* d_ws, size_t ws_size,
                              hipStream_t stream) {
  const float* Q = (const float*)d_in[0];
  const float* K = (const float*)d_in[1];
  const float* V = (const float*)d_in[2];
  const unsigned char* mask = (const unsigned char*)d_in[3];
  float* out = (float*)d_out;

  // 32*8 (b,h) pairs * 16 wave-groups = 4096 waves; 8 waves/block -> 512 blocks
  dim3 grid(512), block(WPB * 32);
  attn_tanh_kernel<<<grid, block, 0, stream>>>(Q, K, V, mask, out);
}
